// SelfAttention_7249904796036
// MI455X (gfx1250) — compile-verified, matrix operand order checked by reading, not run on hardware
//
#include <hip/hip_runtime.h>

// ---------------------------------------------------------------------------
// Self-attention forward for MI455X (gfx1250), bf16 WMMA everywhere.
// B=4, T=2048, D=1024, H=16, HD=64
// Data movement: async global->LDS copies (ASYNCcnt) double-buffered under WMMA.
// ---------------------------------------------------------------------------

typedef __bf16 bf16_t;
typedef __attribute__((ext_vector_type(8)))  __bf16 v8bf;
typedef __attribute__((ext_vector_type(16))) __bf16 v16bf;
typedef __attribute__((ext_vector_type(8)))  float  v8f;

#define BATCH  4
#define SEQ    2048
#define DMODEL 1024
#define NHEAD  16
#define HDIM   64
#define MROWS  (BATCH * SEQ)   // 8192

// ---- helpers ---------------------------------------------------------------

__device__ __forceinline__ bf16_t f2bf(float f) {
  unsigned u = __builtin_bit_cast(unsigned, f);
  unsigned r = (u + 0x7FFFu + ((u >> 16) & 1u)) >> 16;   // round-nearest-even
  unsigned short s = (unsigned short)r;
  return __builtin_bit_cast(bf16_t, s);
}

__device__ __forceinline__ v8f zero8() {
  v8f z = {0.f, 0.f, 0.f, 0.f, 0.f, 0.f, 0.f, 0.f};
  return z;
}

__device__ __forceinline__ v8f wmma_bf16(v16bf a, v16bf b, v8f c) {
  return __builtin_amdgcn_wmma_f32_16x16x32_bf16(
      /*neg_a=*/false, a, /*neg_b=*/false, b,
      /*c_mod=*/(short)0, c, /*reuse_a=*/false, /*reuse_b=*/false);
}

// Generic->LDS offset: LDS aperture keeps the workgroup-relative byte offset
// in the low 32 bits of the generic address (ISA 10.2 aperture mapping).
__device__ __forceinline__ unsigned lds_addr(const void* p) {
  return (unsigned)(uintptr_t)p;
}

// 16B/lane async DMA global -> LDS (VGLOBAL GLOBAL_LOAD_ASYNC_TO_LDS_B128,
// tracked with ASYNCcnt; completes without touching VGPRs).
__device__ __forceinline__ void async_copy16(unsigned lds_off, const void* gptr) {
  asm volatile("global_load_async_to_lds_b128 %0, %1, off"
               :: "v"(lds_off), "v"(gptr) : "memory");
}
__device__ __forceinline__ void async_wait0() {
  asm volatile("s_wait_asynccnt 0x0" ::: "memory");
}

// Load a 16x32 bf16 WMMA operand fragment from a row-major [row][k] tile.
// ISA layout: lanes 0-15 hold row=lane, k in {0..7, 16..23};
//             lanes 16-31 hold row=lane-16, k in {8..15, 24..31}.
// Works for A (row = M) and for B when B's column n is stored as tile row n.
__device__ __forceinline__ v16bf load_frag(const bf16_t* base, int rstride, int lane) {
  const int r  = lane & 15;
  const int kh = (lane >> 4) & 1;
  const bf16_t* p = base + r * rstride + kh * 8;
  v8bf lo = *(const v8bf*)(p);        // k offsets kh*8 + 0..7   (16B aligned)
  v8bf hi = *(const v8bf*)(p + 16);   // k offsets kh*8 + 16..23 (16B aligned)
  v16bf f;
#pragma unroll
  for (int i = 0; i < 8; ++i) { f[i] = lo[i]; f[i + 8] = hi[i]; }
  return f;
}

// Row-wise reductions inside each 16-lane half (C-layout rows live across
// lanes 0-15 and 16-31 separately; xor masks 1,2,4,8 never cross halves).
__device__ __forceinline__ float redmax16(float v) {
#pragma unroll
  for (int m = 8; m; m >>= 1) v = fmaxf(v, __shfl_xor(v, m, 32));
  return v;
}
__device__ __forceinline__ float redsum16(float v) {
#pragma unroll
  for (int m = 8; m; m >>= 1) v += __shfl_xor(v, m, 32);
  return v;
}

// ---- fp32 -> bf16 conversion -----------------------------------------------

__global__ void cvt_f32_bf16(const float* __restrict__ src,
                             bf16_t* __restrict__ dst, int n) {
  int i = blockIdx.x * blockDim.x + threadIdx.x;
  const int stride = gridDim.x * blockDim.x;
  for (; i < n; i += stride) dst[i] = f2bf(src[i]);
}

// ---- generic projection GEMM: Y = A[M,K] * W[N,K]^T + bias -----------------
// out_mode 0: bf16 output, head-major [B, H, T, HD] (for Q/K/V)
// out_mode 1: fp32 output, row-major  [M, N]        (final projection)
// Workgroup: 256 threads (8 waves), 128x128 tile, K-step 32, LDS double-buffer
// filled by async global->LDS DMA overlapped with WMMA on the other buffer.
// Wave grid 4(M) x 2(N); each wave owns 32x64 = 2x4 WMMA accumulators.

__global__ __launch_bounds__(256) void gemm_proj(
    const bf16_t* __restrict__ A, const bf16_t* __restrict__ W,
    const float* __restrict__ bias,
    bf16_t* __restrict__ Ybf, float* __restrict__ Yf,
    int M, int N, int K, int out_mode) {
  __shared__ bf16_t As[2][128][32];   // 8KB x2
  __shared__ bf16_t Bs[2][128][32];   // 8KB x2

  const int tid  = threadIdx.x;
  const int lane = tid & 31;
  const int wave = tid >> 5;
  const int mw = wave & 3;            // wave position along M (0..3)
  const int nw = wave >> 2;           // wave position along N (0..1)
  const int m0 = blockIdx.y * 128;
  const int n0 = blockIdx.x * 128;

  const int sr = tid >> 1;            // staging: row 0..127
  const int sc = (tid & 1) * 16;      // staging: col 0 / 16

  v8f acc[2][4];
#pragma unroll
  for (int i = 0; i < 2; ++i)
#pragma unroll
    for (int j = 0; j < 4; ++j) acc[i][j] = zero8();

  const int KT = K / 32;

  {  // async-prefetch tile 0
    const bf16_t* ga = A + (size_t)(m0 + sr) * K + sc;
    const bf16_t* gw = W + (size_t)(n0 + sr) * K + sc;
    async_copy16(lds_addr(&As[0][sr][sc]),     ga);
    async_copy16(lds_addr(&As[0][sr][sc + 8]), ga + 8);
    async_copy16(lds_addr(&Bs[0][sr][sc]),     gw);
    async_copy16(lds_addr(&Bs[0][sr][sc + 8]), gw + 8);
  }
  async_wait0();
  __syncthreads();

  for (int kt = 0; kt < KT; ++kt) {
    const int cur = kt & 1;
    if (kt + 1 < KT) {   // async-stage next K-slice into the other buffer
      const int k1 = (kt + 1) * 32;
      const int nxt = cur ^ 1;
      const bf16_t* ga = A + (size_t)(m0 + sr) * K + k1 + sc;
      const bf16_t* gw = W + (size_t)(n0 + sr) * K + k1 + sc;
      async_copy16(lds_addr(&As[nxt][sr][sc]),     ga);
      async_copy16(lds_addr(&As[nxt][sr][sc + 8]), ga + 8);
      async_copy16(lds_addr(&Bs[nxt][sr][sc]),     gw);
      async_copy16(lds_addr(&Bs[nxt][sr][sc + 8]), gw + 8);
    }
    v16bf af[2], wf[4];
#pragma unroll
    for (int mi = 0; mi < 2; ++mi)
      af[mi] = load_frag(&As[cur][mw * 32 + mi * 16][0], 32, lane);
#pragma unroll
    for (int ni = 0; ni < 4; ++ni)
      wf[ni] = load_frag(&Bs[cur][nw * 64 + ni * 16][0], 32, lane);
#pragma unroll
    for (int mi = 0; mi < 2; ++mi)
#pragma unroll
      for (int ni = 0; ni < 4; ++ni)
        acc[mi][ni] = wmma_bf16(af[mi], wf[ni], acc[mi][ni]);
    async_wait0();        // DMA for next buffer done (overlapped with WMMA)
    __syncthreads();
  }

  // epilogue: C layout -> lane lr holds col, v + 8*lh is row
  const int lr = lane & 15;
  const int lh = lane >> 4;
#pragma unroll
  for (int mi = 0; mi < 2; ++mi) {
#pragma unroll
    for (int ni = 0; ni < 4; ++ni) {
      const int n = n0 + nw * 64 + ni * 16 + lr;
      const float bv = bias[n];
#pragma unroll
      for (int v = 0; v < 8; ++v) {
        const int m = m0 + mw * 32 + mi * 16 + v + 8 * lh;
        const float val = acc[mi][ni][v] + bv;
        if (out_mode == 0) {
          const int b = m >> 11, t = m & (SEQ - 1);
          const int h = n >> 6,  hd = n & (HDIM - 1);
          Ybf[(((size_t)(b * NHEAD + h) * SEQ + t) << 6) + hd] = f2bf(val);
        } else {
          Yf[(size_t)m * N + n] = val;
        }
      }
    }
  }
}

// ---- flash attention: softmax(QK^T * s + causal) @ V -----------------------
// Grid: (T/128, B*H). 8 waves x 16 query rows. 64-key tiles.

__global__ __launch_bounds__(256) void attn_fwd(
    const bf16_t* __restrict__ Q, const bf16_t* __restrict__ Kg,
    const bf16_t* __restrict__ V, bf16_t* __restrict__ O) {
  __shared__ bf16_t Ks[64][64];       // K tile   [key][hd]      8KB
  __shared__ bf16_t Vt[64][64];       // V tile^T [hd][key]      8KB
  __shared__ bf16_t Pw[8][16][64];    // per-wave P staging     16KB

  const int tid   = threadIdx.x;
  const int lane  = tid & 31;
  const int wave  = tid >> 5;
  const int bh    = blockIdx.y;                 // b*NHEAD + h
  const int qbase = blockIdx.x * 128;
  const int qrow0 = qbase + wave * 16;
  const size_t hb = (size_t)bh * SEQ * HDIM;

  // Q fragments for this wave's 16 rows (two K-steps over HD=64)
  v16bf qf[2];
  qf[0] = load_frag(Q + hb + (size_t)qrow0 * HDIM,      HDIM, lane);
  qf[1] = load_frag(Q + hb + (size_t)qrow0 * HDIM + 32, HDIM, lane);

  v8f oacc[4];
#pragma unroll
  for (int i = 0; i < 4; ++i) oacc[i] = zero8();
  float mrow[8], lrow[8];
#pragma unroll
  for (int v = 0; v < 8; ++v) { mrow[v] = -1e30f; lrow[v] = 0.f; }

  const float scale = 0.125f;   // 1/sqrt(64)
  const int lr = lane & 15, lh = lane >> 4;

  for (int s0 = 0; s0 < qbase + 128; s0 += 64) {   // causal key-tile bound
    {  // stage K tile [key][hd] via async DMA
      const int r = tid >> 2, c = (tid & 3) * 16;
      const bf16_t* g = Kg + hb + (size_t)(s0 + r) * HDIM + c;
      async_copy16(lds_addr(&Ks[r][c]),     g);
      async_copy16(lds_addr(&Ks[r][c + 8]), g + 8);
    }
    {  // stage V transposed [hd][key] (element transpose -> manual path)
      const int key = tid & 63, h0 = (tid >> 6) * 16;
      const bf16_t* g = V + hb + (size_t)(s0 + key) * HDIM + h0;
      v8bf a = *(const v8bf*)g;
      v8bf b = *(const v8bf*)(g + 8);
#pragma unroll
      for (int i = 0; i < 8; ++i) {
        Vt[h0 + i][key]     = a[i];
        Vt[h0 + 8 + i][key] = b[i];
      }
    }
    async_wait0();
    __syncthreads();

    // S = Q K^T : B-column n = key row n of Ks
    v8f sacc[4];
#pragma unroll
    for (int i = 0; i < 4; ++i) sacc[i] = zero8();
#pragma unroll
    for (int ni = 0; ni < 4; ++ni)
#pragma unroll
      for (int ks = 0; ks < 2; ++ks) {
        v16bf kb = load_frag(&Ks[ni * 16][ks * 32], 64, lane);
        sacc[ni] = wmma_bf16(qf[ks], kb, sacc[ni]);
      }

    // causal mask + online softmax
    float sv[4][8];
#pragma unroll
    for (int ni = 0; ni < 4; ++ni) {
      const int col = s0 + ni * 16 + lr;
#pragma unroll
      for (int v = 0; v < 8; ++v) {
        const int row = qrow0 + v + 8 * lh;
        sv[ni][v] = (col > row) ? -1e30f : sacc[ni][v] * scale;
      }
    }
#pragma unroll
    for (int v = 0; v < 8; ++v) {
      float t = fmaxf(fmaxf(sv[0][v], sv[1][v]), fmaxf(sv[2][v], sv[3][v]));
      t = redmax16(t);
      const float mnew  = fmaxf(mrow[v], t);
      const float alpha = __expf(mrow[v] - mnew);
      mrow[v] = mnew;
      float ps = 0.f;
#pragma unroll
      for (int ni = 0; ni < 4; ++ni) {
        const float p = __expf(sv[ni][v] - mnew);
        sv[ni][v] = p;
        ps += p;
        oacc[ni][v] *= alpha;
      }
      lrow[v] = lrow[v] * alpha + redsum16(ps);
    }

    // P: C-layout -> A-layout via wave-private LDS (per-wave DS ordering)
#pragma unroll
    for (int ni = 0; ni < 4; ++ni)
#pragma unroll
      for (int v = 0; v < 8; ++v)
        Pw[wave][v + 8 * lh][ni * 16 + lr] = f2bf(sv[ni][v]);

    // O += P @ V : K-dim = keys; B-column n = Vt row (hd = n)
#pragma unroll
    for (int ks = 0; ks < 2; ++ks) {
      v16bf pa = load_frag(&Pw[wave][0][ks * 32], 64, lane);
#pragma unroll
      for (int ni = 0; ni < 4; ++ni) {
        v16bf vb = load_frag(&Vt[ni * 16][ks * 32], 64, lane);
        oacc[ni] = wmma_bf16(pa, vb, oacc[ni]);
      }
    }
    __syncthreads();
  }

  // normalize and write bf16 [b, t, h*64 + hd] (ready for output projection)
  const int b = bh >> 4, h = bh & 15;
#pragma unroll
  for (int v = 0; v < 8; ++v) {
    const float inv = 1.0f / lrow[v];
    const int row = qrow0 + v + 8 * lh;
#pragma unroll
    for (int ni = 0; ni < 4; ++ni) {
      const int col = h * HDIM + ni * 16 + lr;
      O[((size_t)b * SEQ + row) * DMODEL + col] = f2bf(oacc[ni][v] * inv);
    }
  }
}

// ---------------------------------------------------------------------------

extern "C" void kernel_launch(void* const* d_in, const int* in_sizes, int n_in,
                              void* d_out, int out_size, void* d_ws, size_t ws_size,
                              hipStream_t stream) {
  const float* x  = (const float*)d_in[0];
  const float* wq = (const float*)d_in[1];
  const float* bq = (const float*)d_in[2];
  const float* wk = (const float*)d_in[3];
  const float* bk = (const float*)d_in[4];
  const float* wv = (const float*)d_in[5];
  const float* bv = (const float*)d_in[6];
  const float* wo = (const float*)d_in[7];
  const float* bo = (const float*)d_in[8];
  // d_in[9] is the causal mask tensor; the kernel computes causality
  // analytically (col > row), avoiding T*T mask traffic entirely.

  char* ws = (char*)d_ws;
  size_t off = 0;
  auto take = [&](size_t bytes) -> void* {
    void* p = ws + off;
    off += (bytes + 255) & ~(size_t)255;
    return p;
  };
  const size_t NX = (size_t)MROWS * DMODEL;    // activation elements
  const size_t NW = (size_t)DMODEL * DMODEL;   // weight elements

  bf16_t* xb  = (bf16_t*)take(NX * 2);
  bf16_t* wqb = (bf16_t*)take(NW * 2);
  bf16_t* wkb = (bf16_t*)take(NW * 2);
  bf16_t* wvb = (bf16_t*)take(NW * 2);
  bf16_t* wob = (bf16_t*)take(NW * 2);
  bf16_t* Qb  = (bf16_t*)take(NX * 2);   // [B,H,T,HD]
  bf16_t* Kb  = (bf16_t*)take(NX * 2);   // [B,H,T,HD]
  bf16_t* Vb  = (bf16_t*)take(NX * 2);   // [B,H,T,HD]
  bf16_t* Ab  = (bf16_t*)take(NX * 2);   // attention out, [B,T,D]
  (void)ws_size; (void)in_sizes; (void)n_in; (void)out_size;

  dim3 blk(256);
  cvt_f32_bf16<<<4096, blk, 0, stream>>>(x,  xb,  (int)NX);
  cvt_f32_bf16<<<1024, blk, 0, stream>>>(wq, wqb, (int)NW);
  cvt_f32_bf16<<<1024, blk, 0, stream>>>(wk, wkb, (int)NW);
  cvt_f32_bf16<<<1024, blk, 0, stream>>>(wv, wvb, (int)NW);
  cvt_f32_bf16<<<1024, blk, 0, stream>>>(wo, wob, (int)NW);

  dim3 gg(DMODEL / 128, MROWS / 128);          // (8, 64)
  gemm_proj<<<gg, blk, 0, stream>>>(xb, wqb, bq, Qb, nullptr,
                                    MROWS, DMODEL, DMODEL, 0);
  gemm_proj<<<gg, blk, 0, stream>>>(xb, wkb, bk, Kb, nullptr,
                                    MROWS, DMODEL, DMODEL, 0);
  gemm_proj<<<gg, blk, 0, stream>>>(xb, wvb, bv, Vb, nullptr,
                                    MROWS, DMODEL, DMODEL, 0);

  dim3 ga(SEQ / 128, BATCH * NHEAD);           // (16, 64)
  attn_fwd<<<ga, blk, 0, stream>>>(Qb, Kb, Vb, Ab);

  gemm_proj<<<gg, blk, 0, stream>>>(Ab, wob, bo, nullptr, (float*)d_out,
                                    MROWS, DMODEL, DMODEL, 1);
}
